// CRFLoss_20873541058998
// MI455X (gfx1250) — matrix operand order repaired and checked by reading, not compiled, not run on hardware
//
#include <hip/hip_runtime.h>

typedef __attribute__((ext_vector_type(16))) _Float16 v16h;
typedef __attribute__((ext_vector_type(8)))  _Float16 v8h;
typedef __attribute__((ext_vector_type(8)))  float    v8f;
typedef __attribute__((ext_vector_type(4)))  float    v4f;

#define BATCH   1024
#define S_LEN   512
#define T_TAGS  64
#define AP      68   // alphas row stride in floats (bank-conflict padding)
#define PP      72   // pA row stride in halves (16B-aligned, padded)

#define L2E 1.4426950408889634f   // log2(e)
#define LN2 0.6931471805599453f   // ln(2)

// One block = 16 batch rows. 4 waves: wave w owns output j-tile w (16 states)
// and k-quarter w for the max/exp phases. Recursion is kept in base-2
// (alpha2 = alpha*log2e) so the scan uses raw v_exp_f32/v_log_f32 with no
// range-fixup VALU. Per time step:
//   (pipelined) issue x_{t+1}/mask_{t+1} global loads (index-clamped)
//   phase1: partial row-max of alpha2 over this wave's k-quarter
//   phase2: full row-max m2; p = exp2(alpha2 - m2) -> f16 into LDS (A operand)
//   phase3: two chained v_wmma_f32_16x16x32_f16 (K = 64)
//   phase4: alpha2_new = fma(x, log2e, m2 + log2(acc)); a single combined
//           mask test guards the 8 stores (all-false masks -> one skipped
//           branch, straight-line stores, no old-value loads)
__global__ __launch_bounds__(128) void crf_scan_kernel(
    const float* __restrict__ inputs,
    const float* __restrict__ trans,
    const unsigned char* __restrict__ masks,
    float* __restrict__ log_norm)
{
  __shared__ __align__(16) float     alphas[16 * AP];   // base-2 scaled
  __shared__ __align__(16) _Float16  pA[16 * PP];
  __shared__ float pmax[8 * 16];   // [wave*2+hs][row]
  __shared__ __align__(16) float mrow[16];

  const int tid   = threadIdx.x;
  const int lane  = tid & 31;
  const int wave  = tid >> 5;       // 0..3
  const int row   = lane & 15;
  const int hs    = lane >> 4;      // half-select within wave32
  const int bbase = blockIdx.x * 16;

  // ---- B fragments: E2 = 2^(transitions*log2e), constant across the scan.
  // v_wmma_f32_16x16x32_f16 B layout: lanes 0-15 hold K=0..7 then 16..23,
  // lanes 16-31 hold K=8..15 then 24..31; N = lane%16. Two fragments cover K=64.
  const int col = wave * 16 + row;  // output state j (N)
  v16h bf0, bf1;
  #pragma unroll
  for (int i = 0; i < 16; ++i) {
    const int k0 = hs * 8 + (i < 8 ? i : i + 8);          // K in [0,32)
    const int k1 = 32 + hs * 8 + (i < 8 ? i : i + 8);     // K in [32,64)
    bf0[i] = (_Float16)__builtin_amdgcn_exp2f(trans[k0 * T_TAGS + col] * L2E);
    bf1[i] = (_Float16)__builtin_amdgcn_exp2f(trans[k1 * T_TAGS + col] * L2E);
  }

  // ---- alphas0 = inputs[:, 0, :] * log2e
  for (int e = tid; e < 16 * T_TAGS; e += 128) {
    const int r = e >> 6, jj = e & 63;
    alphas[r * AP + jj] =
        inputs[(size_t)(bbase + r) * S_LEN * T_TAGS + jj] * L2E;
  }
  __syncthreads();

  const int kq = wave * 16 + hs * 8;     // this lane's 8-wide k-chunk
  const int j  = wave * 16 + row;        // this lane's output state (phase 4)

  // Per-row base pointers for the pipelined x_t / mask_t gathers.
  const float*         px[8];
  const unsigned char* pm[8];
  #pragma unroll
  for (int v = 0; v < 8; ++v) {
    const int b = bbase + v + hs * 8;
    px[v] = inputs + (size_t)b * S_LEN * T_TAGS + j;
    pm[v] = masks + (size_t)b * S_LEN;
  }

  // Preload x_t / mask_t for t = 1.
  float         xv[8];
  unsigned char mb[8];
  #pragma unroll
  for (int v = 0; v < 8; ++v) { xv[v] = px[v][(size_t)1 * T_TAGS]; mb[v] = pm[v][1]; }

  for (int t = 1; t < S_LEN; ++t) {
    // -------- pipelined prefetch of x_{t+1}, mask_{t+1} (clamped) --------
    const int tn = (t < S_LEN - 1) ? t + 1 : t;   // last iter: dummy reload
    float         xn[8];
    unsigned char mn[8];
    #pragma unroll
    for (int v = 0; v < 8; ++v) {
      xn[v] = px[v][(size_t)tn * T_TAGS];
      mn[v] = pm[v][tn];
    }

    // -------- phase 1: partial row max over k-quarter --------
    const v4f a0 = *(const v4f*)&alphas[row * AP + kq];
    const v4f a1 = *(const v4f*)&alphas[row * AP + kq + 4];
    float mx = fmaxf(fmaxf(fmaxf(a0[0], a0[1]), fmaxf(a0[2], a0[3])),
                     fmaxf(fmaxf(a1[0], a1[1]), fmaxf(a1[2], a1[3])));
    pmax[(wave * 2 + hs) * 16 + row] = mx;
    __syncthreads();

    // -------- phase 2: full row max; p = 2^(a2 - m2) -> f16 A operand ----
    float m = pmax[row];
    #pragma unroll
    for (int q = 1; q < 8; ++q) m = fmaxf(m, pmax[q * 16 + row]);
    v8h ph;
    #pragma unroll
    for (int i = 0; i < 4; ++i)
      ph[i]     = (_Float16)__builtin_amdgcn_exp2f(a0[i] - m);
    #pragma unroll
    for (int i = 0; i < 4; ++i)
      ph[4 + i] = (_Float16)__builtin_amdgcn_exp2f(a1[i] - m);
    *(v8h*)&pA[row * PP + kq] = ph;
    if (tid < 16) mrow[tid] = m;   // wave 0: lane==row, full row max
    __syncthreads();

    // -------- phase 3: load A fragments, two chained WMMAs (K=64) -------
    // A layout (16x32 f16): lanes 0-15: K=base+0..7 then base+16..23;
    //                       lanes 16-31: K=base+8..15 then base+24..31.
    const v8h lo0 = *(const v8h*)&pA[row * PP +  0 + hs * 8];
    const v8h hi0 = *(const v8h*)&pA[row * PP + 16 + hs * 8];
    const v8h lo1 = *(const v8h*)&pA[row * PP + 32 + hs * 8];
    const v8h hi1 = *(const v8h*)&pA[row * PP + 48 + hs * 8];
    v16h af0, af1;
    #pragma unroll
    for (int i = 0; i < 8; ++i) {
      af0[i] = lo0[i]; af0[8 + i] = hi0[i];
      af1[i] = lo1[i]; af1[8 + i] = hi1[i];
    }
    v8f c = {};
    c = __builtin_amdgcn_wmma_f32_16x16x32_f16(false, af0, false, bf0,
                                               (short)0, c, false, false);
    c = __builtin_amdgcn_wmma_f32_16x16x32_f16(false, af1, false, bf1,
                                               (short)0, c, false, false);

    // -------- phase 4: alpha2_new = fma(x,log2e, m2+log2(c)) -------------
    // D layout: VGPR v -> M = v + hs*8, N = lane%16.
    const v4f mr0 = *(const v4f*)&mrow[hs * 8];      // m2 for M = hs*8..+3
    const v4f mr1 = *(const v4f*)&mrow[hs * 8 + 4];  // m2 for M = hs*8+4..+7
    float na[8];
    #pragma unroll
    for (int v = 0; v < 8; ++v) {
      const float mv = (v < 4) ? mr0[v] : mr1[v - 4];
      na[v] = fmaf(xv[v], L2E, mv + __builtin_amdgcn_logf(c[v]));
    }
    const unsigned anym = (unsigned)mb[0] | mb[1] | mb[2] | mb[3] |
                          mb[4] | mb[5] | mb[6] | mb[7];
    if (anym == 0) {
      // common case: straight-line stores, no old-value traffic
      #pragma unroll
      for (int v = 0; v < 8; ++v)
        alphas[(v + hs * 8) * AP + j] = na[v];
    } else {
      // rare: masked rows keep their previous alpha
      #pragma unroll
      for (int v = 0; v < 8; ++v) {
        const float old = alphas[(v + hs * 8) * AP + j];
        alphas[(v + hs * 8) * AP + j] = mb[v] ? old : na[v];
      }
    }
    __syncthreads();

    #pragma unroll
    for (int v = 0; v < 8; ++v) { xv[v] = xn[v]; mb[v] = mn[v]; }
  }

  // ---- final log_norm[b] = ln2 * log2-sum-exp2_j(alpha2[b, :]) ----
  if (tid < 16) {
    const int r = tid;
    float mm = alphas[r * AP];
    for (int j2 = 1; j2 < T_TAGS; ++j2) mm = fmaxf(mm, alphas[r * AP + j2]);
    float s = 0.0f;
    for (int j2 = 0; j2 < T_TAGS; ++j2)
      s += __builtin_amdgcn_exp2f(alphas[r * AP + j2] - mm);
    log_norm[bbase + r] = LN2 * (mm + __builtin_amdgcn_logf(s));
  }
}

// unary + binary gather-sum per batch
__global__ __launch_bounds__(256) void crf_ub_kernel(
    const float* __restrict__ inputs,
    const float* __restrict__ trans,
    const unsigned char* __restrict__ masks,
    const int* __restrict__ tags,
    float* __restrict__ ub)
{
  const int b = blockIdx.x * blockDim.x + threadIdx.x;
  if (b >= BATCH) return;
  float acc = 0.0f;
  int prev = 0;
  for (int s = 0; s < S_LEN; ++s) {
    const int tg = tags[(size_t)b * S_LEN + s];
    const unsigned char mb = masks[(size_t)b * S_LEN + s];
    if (!mb) {
      acc += inputs[((size_t)b * S_LEN + s) * T_TAGS + tg];
      if (s > 0) acc += trans[prev * T_TAGS + tg];
    }
    prev = tg;
  }
  ub[b] = acc;
}

// loss = sum_b(log_norm - unary - binary)/B ; also emit transitions (output #2)
__global__ __launch_bounds__(256) void crf_finalize_kernel(
    const float* __restrict__ log_norm,
    const float* __restrict__ ub,
    const float* __restrict__ trans,
    float* __restrict__ out)
{
  __shared__ float red[256];
  const int tid = threadIdx.x;
  float acc = 0.0f;
  for (int b = tid; b < BATCH; b += 256) acc += log_norm[b] - ub[b];
  red[tid] = acc;
  __syncthreads();
  for (int s = 128; s > 0; s >>= 1) {
    if (tid < s) red[tid] += red[tid + s];
    __syncthreads();
  }
  if (tid == 0) out[0] = red[0] / (float)BATCH;
  for (int i = tid; i < T_TAGS * T_TAGS; i += 256) out[1 + i] = trans[i];
}

extern "C" void kernel_launch(void* const* d_in, const int* in_sizes, int n_in,
                              void* d_out, int out_size, void* d_ws, size_t ws_size,
                              hipStream_t stream) {
  const float* inputs         = (const float*)d_in[0];
  const float* trans          = (const float*)d_in[1];
  const unsigned char* masks  = (const unsigned char*)d_in[2];
  const int* tags             = (const int*)d_in[3];
  float* out = (float*)d_out;

  float* log_norm = (float*)d_ws;        // 1024 floats
  float* ub       = log_norm + BATCH;    // 1024 floats

  crf_scan_kernel<<<BATCH / 16, 128, 0, stream>>>(inputs, trans, masks, log_norm);
  crf_ub_kernel<<<BATCH / 256, 256, 0, stream>>>(inputs, trans, masks, tags, ub);
  crf_finalize_kernel<<<1, 256, 0, stream>>>(log_norm, ub, trans, out);
}
